// ScatterSelfAttention_13554916786671
// MI455X (gfx1250) — matrix operand order repaired and checked by reading, not compiled, not run on hardware
//
#include <hip/hip_runtime.h>

// ---------------------------------------------------------------------------
// ScatterSelfAttention for MI455X (gfx1250, wave32, WMMA bf16 16x16x32)
//   N = 65536 rows, DIM = 512, NUM_SEGMENTS = 256, MAX_SEG_LEN = 512
// Pipeline:
//   1) segment counts + exclusive scan (indices are sorted -> rows contiguous)
//   2) fp32 -> bf16 conversion of x and the three weight matrices
//   3) projection: 8-wave workgroups, weights staged per-K-step in LDS,
//      q/k row-major bf16, v stored transposed (d-major) for P@V B-fragments
//   4) per-(segment, 16-row q-tile) attention: scores -> softmax -> P@V
// ---------------------------------------------------------------------------

#define N_TOT 65536
#define DIMF  512
#define NSEG  256

typedef __attribute__((ext_vector_type(16))) __bf16 bf16x16;
typedef __attribute__((ext_vector_type(8)))  __bf16 bf16x8;
typedef __attribute__((ext_vector_type(8)))  float  f32x8;

union AFrag { bf16x16 v; bf16x8 h[2]; };

__device__ __forceinline__ __bf16 f2bf(float f) {
  unsigned u = __builtin_bit_cast(unsigned, f);
  unsigned r = (u + 0x7FFFu + ((u >> 16) & 1u)) >> 16;   // round-to-nearest-even
  unsigned short s = (unsigned short)r;
  return __builtin_bit_cast(__bf16, s);
}

__device__ __forceinline__ f32x8 wmma_bf16(bf16x16 a, bf16x16 b, f32x8 c) {
  // D = A(16x32 bf16) * B(32x16 bf16) + C(16x16 f32)
  return __builtin_amdgcn_wmma_f32_16x16x32_bf16(false, a, false, b,
                                                 (short)0, c, false, false);
}

// ---------------------------------------------------------------- bookkeeping
__global__ void zero_counts_kernel(int* counts) { counts[threadIdx.x] = 0; }

__global__ void count_kernel(const long long* __restrict__ idx,
                             int* __restrict__ counts) {
  int n = blockIdx.x * blockDim.x + threadIdx.x;
  int v = (int)idx[n];
  if (v < 0) v = 0; if (v >= NSEG) v = NSEG - 1;
  atomicAdd(&counts[v], 1);
}

__global__ void scan_kernel(const int* __restrict__ counts,
                            int* __restrict__ starts) {
  if (threadIdx.x == 0) {
    int run = 0;
    for (int i = 0; i < NSEG; ++i) { starts[i] = run; run += counts[i]; }
  }
}

// ------------------------------------------------------------- f32 -> bf16
__global__ void cvt_kernel(const float* __restrict__ in,
                           __bf16* __restrict__ out) {
  size_t i = (size_t)blockIdx.x * blockDim.x + threadIdx.x;
  float4 f = ((const float4*)in)[i];
  union { unsigned long long u; __bf16 b[4]; } pk;
  pk.b[0] = f2bf(f.x); pk.b[1] = f2bf(f.y);
  pk.b[2] = f2bf(f.z); pk.b[3] = f2bf(f.w);
  *(unsigned long long*)(out + 4 * i) = pk.u;
}

// ------------------------------------------------------------- projections
// 8 waves per workgroup, each wave owns a 16-row x 512-col strip of
// y = x @ W.T (128 rows per workgroup). The 512x32 bf16 weight slab for the
// current K-step is staged cooperatively in LDS and shared by all 8 waves,
// cutting weight re-fetch traffic 8x vs per-wave global loads.
// which: 0 -> Q (row-major), 1 -> K (row-major), 2 -> V (stored transposed).
#define SLAB_W 40   // 32 halves + 8 pad: 80B row stride, 16B aligned, few conflicts
__global__ __launch_bounds__(256) void proj_kernel(
    const __bf16* __restrict__ xb, const __bf16* __restrict__ wball,
    __bf16* __restrict__ qb, __bf16* __restrict__ kb,
    __bf16* __restrict__ vt) {
  __shared__ __attribute__((aligned(16))) __bf16 slab[DIMF][SLAB_W];

  const int which = blockIdx.y;
  const int tid   = threadIdx.x;
  const int wave  = tid >> 5;
  const int lane  = tid & 31;
  const int hi    = lane >> 4;
  const int m     = lane & 15;
  const int r0    = blockIdx.x * 128 + wave * 16;
  const __bf16* W = wball + (size_t)which * DIMF * DIMF;

  f32x8 acc[32] = {};
  const __bf16* arow = xb + (size_t)(r0 + m) * DIMF;

  for (int st = 0; st < 16; ++st) {               // K loop over d_in, 32/step
    __syncthreads();                              // slab free from prev step
    // ---- stage W[:, st*32 .. st*32+31] into LDS: 2 rows x 64B per thread
#pragma unroll
    for (int r = 0; r < 2; ++r) {
      const int row = tid + r * 256;
      const uint4* src = (const uint4*)(W + (size_t)row * DIMF + st * 32);
      uint4* dst = (uint4*)(&slab[row][0]);
      dst[0] = src[0]; dst[1] = src[1]; dst[2] = src[2]; dst[3] = src[3];
    }
    __syncthreads();

    AFrag a;
    a.h[0] = *(const bf16x8*)(arow + st * 32 + 16 * hi);
    a.h[1] = *(const bf16x8*)(arow + st * 32 + 16 * hi + 8);
#pragma unroll
    for (int nt = 0; nt < 32; ++nt) {             // 32 output col tiles
      bf16x16 b = *(const bf16x16*)(&slab[nt * 16 + m][16 * hi]);
      acc[nt] = wmma_bf16(a.v, b, acc[nt]);
    }
  }

  if (which == 2) {
    // V transposed: lane owns column d = nt*16+m, rows r0+8*hi .. +7 contiguous
#pragma unroll
    for (int nt = 0; nt < 32; ++nt) {
      union { uint4 u; __bf16 b[8]; } pk;
#pragma unroll
      for (int v = 0; v < 8; ++v) pk.b[v] = f2bf(acc[nt][v]);
      *(uint4*)(vt + (size_t)(nt * 16 + m) * N_TOT + r0 + 8 * hi) = pk.u;
    }
  } else {
    __bf16* dst = (which == 0) ? qb : kb;
#pragma unroll
    for (int nt = 0; nt < 32; ++nt)
#pragma unroll
      for (int v = 0; v < 8; ++v)
        dst[(size_t)(r0 + v + 8 * hi) * DIMF + nt * 16 + m] = f2bf(acc[nt][v]);
  }
}

// ------------------------------------------------------------- attention
// One wave per (segment, 16-row query tile). Two-pass softmax via LDS.
__global__ __launch_bounds__(32) void attn_kernel(
    const __bf16* __restrict__ qbuf, const __bf16* __restrict__ kbuf,
    const __bf16* __restrict__ vt, const int* __restrict__ counts,
    const int* __restrict__ starts, float* __restrict__ out) {
  __shared__ float sc[16][516];                              // +4 pad: no bank conflicts
  __shared__ __attribute__((aligned(16))) __bf16 pb[16][584]; // shifted P, padded
  __shared__ float redm[2][16];
  __shared__ float reds[2][16];

  const int s  = blockIdx.y;
  const int qt = blockIdx.x;
  int c = counts[s];
  if (c > 512) c = 512;                         // MAX_SEG_LEN cap
  if (qt * 16 >= c) return;                     // uniform early exit
  const int start = starts[s];
  const int lane = threadIdx.x, hi = lane >> 4, m = lane & 15;
  const int r0 = start + qt * 16;

  // ---- load Q fragments for this 16-row tile (16 K-steps over d=512)
  int qr = r0 + m; if (qr > N_TOT - 1) qr = N_TOT - 1;
  const __bf16* qrow = qbuf + (size_t)qr * DIMF;
  AFrag qf[16];
#pragma unroll
  for (int st = 0; st < 16; ++st) {
    qf[st].h[0] = *(const bf16x8*)(qrow + st * 32 + 16 * hi);
    qf[st].h[1] = *(const bf16x8*)(qrow + st * 32 + 16 * hi + 8);
  }

  // ---- scores: S = (Q K^T) * 1/sqrt(d), masked to -1e9 beyond count
  const int nkt = (c + 15) >> 4;
  const float scale = 0.044194173824159216f;    // 1/sqrt(512)
  for (int j = 0; j < nkt; ++j) {
    int kr = start + j * 16 + m; if (kr > N_TOT - 1) kr = N_TOT - 1;
    const __bf16* krow = kbuf + (size_t)kr * DIMF;
    f32x8 cs = {};
#pragma unroll
    for (int st = 0; st < 16; ++st) {
      bf16x16 b = *(const bf16x16*)(krow + st * 32 + 16 * hi);
      cs = wmma_bf16(qf[st].v, b, cs);
    }
    const int colg = j * 16 + m;
#pragma unroll
    for (int v = 0; v < 8; ++v) {
      float val = cs[v] * scale;
      if (colg >= c) val = -1e9f;
      sc[v + 8 * hi][colg] = val;
    }
  }
  __syncthreads();

  // ---- softmax over row m; half `hi` handles columns of parity hi
  const int cpad = nkt * 16;
  float mx = -3.0e38f;
  for (int col = hi; col < cpad; col += 2) mx = fmaxf(mx, sc[m][col]);
  redm[hi][m] = mx;
  __syncthreads();
  const float mf = fmaxf(redm[0][m], redm[1][m]);
  float sum = 0.f;
  for (int col = hi; col < cpad; col += 2) {
    float e = __expf(sc[m][col] - mf);
    sc[m][col] = e;
    sum += e;
  }
  reds[hi][m] = sum;
  __syncthreads();
  const float inv = 1.f / (reds[0][m] + reds[1][m]);

  // ---- write P, column-shifted so V loads stay 32B aligned
  const int base = start & ~31;
  const int lead = start - base;
  for (int col = hi; col < cpad; col += 2)
    pb[m][lead + col] = f2bf(sc[m][col] * inv);
  const int cpad32 = (lead + cpad + 31) & ~31;
  for (int t = hi; t < lead; t += 2)                 pb[m][t] = f2bf(0.f);
  for (int t = lead + cpad + hi; t < cpad32; t += 2) pb[m][t] = f2bf(0.f);
  __syncthreads();

  // ---- O = P @ V  (V read column-major from Vt, aligned at `base`)
  const int nks = cpad32 >> 5;
  const int rowlim = c - qt * 16;
#pragma unroll 1
  for (int nt = 0; nt < 32; ++nt) {
    f32x8 acc = {};
    const __bf16* vcol = vt + (size_t)(nt * 16 + m) * N_TOT + base;
    for (int ks = 0; ks < nks; ++ks) {
      AFrag p;
      p.h[0] = *(const bf16x8*)(&pb[m][ks * 32 + 16 * hi]);
      p.h[1] = *(const bf16x8*)(&pb[m][ks * 32 + 16 * hi + 8]);
      bf16x16 b = *(const bf16x16*)(vcol + ks * 32 + 16 * hi);
      acc = wmma_bf16(p.v, b, acc);
    }
#pragma unroll
    for (int v = 0; v < 8; ++v) {
      const int r = v + 8 * hi;
      if (r < rowlim)
        out[(size_t)(r0 + r) * DIMF + nt * 16 + m] = acc[v];
    }
  }
}

// ---------------------------------------------------------------- launcher
extern "C" void kernel_launch(void* const* d_in, const int* in_sizes, int n_in,
                              void* d_out, int out_size, void* d_ws, size_t ws_size,
                              hipStream_t stream) {
  const float*     x   = (const float*)d_in[0];
  const long long* idx = (const long long*)d_in[1];
  const float*     Kw  = (const float*)d_in[2];
  const float*     Qw  = (const float*)d_in[3];
  const float*     Vw  = (const float*)d_in[4];
  float* out = (float*)d_out;

  const size_t NB = (size_t)N_TOT * DIMF;        // elements per [N, DIM] buffer
  __bf16* xb = (__bf16*)d_ws;                    //  64 MB
  __bf16* qb = xb + NB;                          //  64 MB
  __bf16* kb = qb + NB;                          //  64 MB
  __bf16* vt = kb + NB;                          //  64 MB (transposed V)
  __bf16* wb = vt + NB;                          // 1.5 MB (Qw,Kw,Vw bf16)
  int* counts = (int*)(wb + (size_t)3 * DIMF * DIMF);
  int* starts = counts + NSEG;

  zero_counts_kernel<<<1, NSEG, 0, stream>>>(counts);
  count_kernel<<<N_TOT / 256, 256, 0, stream>>>(idx, counts);
  scan_kernel<<<1, 32, 0, stream>>>(counts, starts);

  cvt_kernel<<<(int)(NB / 4 / 256), 256, 0, stream>>>(x, xb);
  cvt_kernel<<<DIMF * DIMF / 4 / 256, 256, 0, stream>>>(Qw, wb);
  cvt_kernel<<<DIMF * DIMF / 4 / 256, 256, 0, stream>>>(Kw, wb + DIMF * DIMF);
  cvt_kernel<<<DIMF * DIMF / 4 / 256, 256, 0, stream>>>(Vw, wb + 2 * DIMF * DIMF);

  proj_kernel<<<dim3(N_TOT / 128, 3), 256, 0, stream>>>(xb, wb, qb, kb, vt);
  attn_kernel<<<dim3(512 / 16, NSEG), 32, 0, stream>>>(qb, kb, vt, counts,
                                                       starts, out);
}
